// LlamaAttention_89120571392383
// MI455X (gfx1250) — matrix operand order repaired and checked by reading, not compiled, not run on hardware
//
#include <hip/hip_runtime.h>

// ---------------------------------------------------------------------------
// LLaMA attention block for gfx1250 (MI455X), bf16 WMMA pipeline.
// All matmuls use v_wmma_f32_16x16x32_bf16 (wave32, 16x16 tiles, K=32).
// ---------------------------------------------------------------------------

typedef __attribute__((ext_vector_type(16))) __bf16 v16bf;
typedef __attribute__((ext_vector_type(8)))  __bf16 v8bf;
typedef __attribute__((ext_vector_type(8)))  float  v8f;

#define SEQ     2048
#define HIDDEN  4096
#define NH      32
#define NKV     8
#define HD      128
#define QKVN    6144   // 4096 q + 1024 k + 1024 v

// ---------------------------------------------------------------------------
// WMMA helpers (CDNA5 wave32 fragment layouts, CDNA5 ISA §7.12.2)
// ---------------------------------------------------------------------------

__device__ __forceinline__ v8f wmma_bf16(v16bf a, v16bf b, v8f c) {
  // 8 args: (neg_a, A, neg_b, B, c_mod, C, reuse_a, reuse_b)
  return __builtin_amdgcn_wmma_f32_16x16x32_bf16(false, a, false, b, (short)0, c,
                                                 false, false);
}

// A fragment, 16x32 bf16 (MxK), row-major source with leading dim `ld`.
// lane L: row M = L&15; half g = L>>4 supplies k = g*8+0..7 and 16+g*8+0..7.
__device__ __forceinline__ v16bf load_a_frag(const __bf16* base, int ld) {
  const int lane = threadIdx.x & 31;
  const int m = lane & 15;
  const int g = lane >> 4;
  const __bf16* p = base + (size_t)m * ld + g * 8;
  v8bf lo = *(const v8bf*)(p);        // k = g*8 + 0..7
  v8bf hi = *(const v8bf*)(p + 16);   // k = 16 + g*8 + 0..7
  return __builtin_shufflevector(lo, hi, 0, 1, 2, 3, 4, 5, 6, 7,
                                 8, 9, 10, 11, 12, 13, 14, 15);
}

// B fragment, 32x16 bf16 (KxN), stored COLUMN-major: column n is 32
// K-contiguous bf16 at base + n*ld.  lane L: col n = L&15, half g = L>>4
// supplies k = g*8 + 0..7 and 16 + g*8 + 0..7.
__device__ __forceinline__ v16bf load_b_frag(const __bf16* base, int ld) {
  const int lane = threadIdx.x & 31;
  const int n = lane & 15;
  const int g = lane >> 4;
  const __bf16* p = base + (size_t)n * ld + g * 8;
  v8bf lo = *(const v8bf*)(p);
  v8bf hi = *(const v8bf*)(p + 16);
  return __builtin_shufflevector(lo, hi, 0, 1, 2, 3, 4, 5, 6, 7,
                                 8, 9, 10, 11, 12, 13, 14, 15);
}

// wave32 intra-row (16-lane half) reductions for the C fragment layout
__device__ __forceinline__ float rowred_max(float x) {
  x = fmaxf(x, __shfl_xor(x, 1, 32));
  x = fmaxf(x, __shfl_xor(x, 2, 32));
  x = fmaxf(x, __shfl_xor(x, 4, 32));
  x = fmaxf(x, __shfl_xor(x, 8, 32));
  return x;
}
__device__ __forceinline__ float rowred_sum(float x) {
  x += __shfl_xor(x, 1, 32);
  x += __shfl_xor(x, 2, 32);
  x += __shfl_xor(x, 4, 32);
  x += __shfl_xor(x, 8, 32);
  return x;
}

// ---------------------------------------------------------------------------
// 1) f32 -> bf16 convert (grid-stride)
// ---------------------------------------------------------------------------
__global__ void cvt_f32_bf16(const float* __restrict__ s, __bf16* __restrict__ d,
                             long long n) {
  long long i = (long long)blockIdx.x * blockDim.x + threadIdx.x;
  long long stride = (long long)gridDim.x * blockDim.x;
  for (; i < n; i += stride) d[i] = (__bf16)s[i];
}

// ---------------------------------------------------------------------------
// 2/5) GEMM  C[M,N] = X[M,K] @ W[N,K]^T   (bf16 in, f32 accum)
// wave tile 32x64, 4 waves per block -> block tile 32x256.
// ---------------------------------------------------------------------------
__global__ __launch_bounds__(128) void gemm_bf16(const __bf16* __restrict__ X,
                                                 const __bf16* __restrict__ W,
                                                 __bf16* __restrict__ Ob,
                                                 float* __restrict__ Of,
                                                 int M, int N, int K) {
  const int wave = threadIdx.x >> 5;
  const int lane = threadIdx.x & 31;
  const int n = lane & 15;
  const int g = lane >> 4;
  const int m0 = blockIdx.x * 32;
  const int n0 = (blockIdx.y * 4 + wave) * 64;

  v8f acc[2][4];
  for (int i = 0; i < 2; ++i)
    for (int j = 0; j < 4; ++j)
      for (int r = 0; r < 8; ++r) acc[i][j][r] = 0.f;

  const __bf16* Xp = X + (size_t)m0 * K;
  for (int k0 = 0; k0 < K; k0 += 32) {
    // prefetch 8 K slabs ahead (lowers to global_prefetch_b8)
    __builtin_prefetch(Xp + k0 + 256, 0, 0);
    v16bf a0 = load_a_frag(Xp + k0, K);
    v16bf a1 = load_a_frag(Xp + (size_t)16 * K + k0, K);
#pragma unroll
    for (int j = 0; j < 4; ++j) {
      v16bf b = load_b_frag(W + (size_t)(n0 + j * 16) * K + k0, K);
      acc[0][j] = wmma_bf16(a0, b, acc[0][j]);
      acc[1][j] = wmma_bf16(a1, b, acc[1][j]);
    }
  }

  // C layout: lane, vgpr r -> (M = r + 8*g, N = lane&15)
  for (int i = 0; i < 2; ++i)
    for (int j = 0; j < 4; ++j)
      for (int r = 0; r < 8; ++r) {
        size_t row = m0 + i * 16 + r + 8 * g;
        size_t col = n0 + j * 16 + n;
        float v = acc[i][j][r];
        if (Ob) Ob[row * N + col] = (__bf16)v;
        else    Of[row * N + col] = v;
      }
}

// ---------------------------------------------------------------------------
// 3) RoPE + layout: qkv[S,6144] bf16 ->
//      qo [H][S][D]   (q * D^-0.5, rotated)
//      ko [HKV][S][D] (rotated)
//      vt [HKV][D][S] (transposed for PV B-fragments)
// one thread per (s, h, i<64) rotation pair
// ---------------------------------------------------------------------------
__global__ void rope_layout(const __bf16* __restrict__ qkv,
                            const int* __restrict__ pos,
                            __bf16* __restrict__ qo,
                            __bf16* __restrict__ ko,
                            __bf16* __restrict__ vt) {
  int t = blockIdx.x * blockDim.x + threadIdx.x;
  if (t >= SEQ * NH * 64) return;
  int i = t & 63;
  int h = (t >> 6) & (NH - 1);
  int s = t >> 11;

  float p = (float)pos[s];
  // theta^(-i/64) = exp(-i * ln(1e4)/64)
  float inv_freq = __expf(-(float)i * (9.210340371976184f / 64.0f));
  float ang = p * inv_freq;
  float c = __cosf(ang), sn = __sinf(ang);
  const float scale = 0.08838834764831845f;  // 128^-0.5

  const __bf16* qr = qkv + (size_t)s * QKVN + h * HD;
  float x1 = (float)qr[i], x2 = (float)qr[i + 64];
  size_t qidx = ((size_t)h * SEQ + s) * HD + i;
  qo[qidx]      = (__bf16)((x1 * c - x2 * sn) * scale);
  qo[qidx + 64] = (__bf16)((x2 * c + x1 * sn) * scale);

  if (h < NKV) {
    const __bf16* kr = qkv + (size_t)s * QKVN + HIDDEN + h * HD;
    float k1 = (float)kr[i], k2 = (float)kr[i + 64];
    size_t kidx = ((size_t)h * SEQ + s) * HD + i;
    ko[kidx]      = (__bf16)(k1 * c - k2 * sn);
    ko[kidx + 64] = (__bf16)(k2 * c + k1 * sn);

    const __bf16* vr = qkv + (size_t)s * QKVN + HIDDEN + NKV * HD + h * HD;
    vt[((size_t)h * HD + i) * SEQ + s]        = vr[i];
    vt[((size_t)h * HD + i + 64) * SEQ + s]   = vr[i + 64];
  }
}

// ---------------------------------------------------------------------------
// 4) causal GQA flash attention: one wave per (head, 16-row q tile).
//    K processed in 32-key tiles; online softmax; P staged through 1KB LDS
//    to convert the C-layout score fragment into an A-layout fragment.
// ---------------------------------------------------------------------------
__global__ __launch_bounds__(32) void attn_kernel(const __bf16* __restrict__ qb,
                                                  const __bf16* __restrict__ kb,
                                                  const __bf16* __restrict__ vt,
                                                  __bf16* __restrict__ ob) {
  __shared__ alignas(16) __bf16 pstage[16 * 32];

  const int qt = blockIdx.x;       // q tile 0..127
  const int h  = blockIdx.y;       // head 0..31
  const int hk = h >> 2;           // GQA group of 4
  const int lane = threadIdx.x & 31;
  const int n = lane & 15;
  const int g = lane >> 4;
  const int qb0 = qt * 16;

  // Q fragments for all of D=128 (4 x K=32)
  v16bf qf[4];
  const __bf16* qptr = qb + ((size_t)h * SEQ + qb0) * HD;
#pragma unroll
  for (int c = 0; c < 4; ++c) qf[c] = load_a_frag(qptr + c * 32, HD);

  float mrow[8], lrow[8];
  v8f acc[8];
  for (int r = 0; r < 8; ++r) { mrow[r] = -1e30f; lrow[r] = 0.f; }
  for (int nd = 0; nd < 8; ++nd)
    for (int r = 0; r < 8; ++r) acc[nd][r] = 0.f;

  const int kend = qb0 + 16;  // causal horizon
  for (int kb0 = 0; kb0 < kend; kb0 += 32) {
    // scores for 32 keys: two 16x16 fragments
    v8f s0, s1;
    for (int r = 0; r < 8; ++r) { s0[r] = 0.f; s1[r] = 0.f; }
    const __bf16* kptr = kb + ((size_t)hk * SEQ + kb0) * HD;
#pragma unroll
    for (int c = 0; c < 4; ++c) {
      v16bf b0 = load_b_frag(kptr + c * 32, HD);
      v16bf b1 = load_b_frag(kptr + (size_t)16 * HD + c * 32, HD);
      s0 = wmma_bf16(qf[c], b0, s0);
      s1 = wmma_bf16(qf[c], b1, s1);
    }

    // causal mask (element row = qb0 + r + 8g, col = kb0 (+16) + n)
#pragma unroll
    for (int r = 0; r < 8; ++r) {
      int row = qb0 + r + 8 * g;
      if (kb0 + n > row)      s0[r] = -1e30f;
      if (kb0 + 16 + n > row) s1[r] = -1e30f;
    }

    // online softmax update
    float sf[8], p0[8], p1[8];
#pragma unroll
    for (int r = 0; r < 8; ++r) {
      float rm = rowred_max(fmaxf(s0[r], s1[r]));
      float mn = fmaxf(mrow[r], rm);
      sf[r] = __expf(mrow[r] - mn);
      mrow[r] = mn;
      p0[r] = __expf(s0[r] - mn);
      p1[r] = __expf(s1[r] - mn);
      float rs = rowred_sum(p0[r] + p1[r]);
      lrow[r] = lrow[r] * sf[r] + rs;
    }
#pragma unroll
    for (int nd = 0; nd < 8; ++nd)
      for (int r = 0; r < 8; ++r) acc[nd][r] *= sf[r];

    // stage P (16x32) to LDS, re-read as an A-layout fragment
    __syncthreads();  // WAR vs previous iteration's reads (1-wave block)
#pragma unroll
    for (int r = 0; r < 8; ++r) {
      int row = r + 8 * g;
      pstage[row * 32 + n]      = (__bf16)p0[r];
      pstage[row * 32 + 16 + n] = (__bf16)p1[r];
    }
    __syncthreads();
    const __bf16* pp = &pstage[(size_t)(lane & 15) * 32 + g * 8];
    v8bf plo = *(const v8bf*)(pp);
    v8bf phi = *(const v8bf*)(pp + 16);
    v16bf pf = __builtin_shufflevector(plo, phi, 0, 1, 2, 3, 4, 5, 6, 7,
                                       8, 9, 10, 11, 12, 13, 14, 15);

    // O += P @ V   (V^T layout -> contiguous B columns)
#pragma unroll
    for (int nd = 0; nd < 8; ++nd) {
      v16bf bv = load_b_frag(vt + ((size_t)hk * HD + nd * 16) * SEQ + kb0, SEQ);
      acc[nd] = wmma_bf16(pf, bv, acc[nd]);
    }
  }

  // normalize and write attn output, layout [s][h*128+d] for the out-proj GEMM
  float inv[8];
  for (int r = 0; r < 8; ++r) inv[r] = 1.0f / lrow[r];
#pragma unroll
  for (int nd = 0; nd < 8; ++nd)
    for (int r = 0; r < 8; ++r) {
      size_t row = qb0 + r + 8 * g;
      ob[row * (size_t)(NH * HD) + h * HD + nd * 16 + n] =
          (__bf16)(acc[nd][r] * inv[r]);
    }
}

// ---------------------------------------------------------------------------
// Launch
// ---------------------------------------------------------------------------
extern "C" void kernel_launch(void* const* d_in, const int* in_sizes, int n_in,
                              void* d_out, int out_size, void* d_ws, size_t ws_size,
                              hipStream_t stream) {
  const float* hs  = (const float*)d_in[0];  // (1,2048,4096)
  const int*   pos = (const int*)d_in[1];    // (2048,)
  // d_in[2] page_indices: identity round-trip in the reference, unused here
  const float* Wq = (const float*)d_in[3];   // (4096,4096)
  const float* Wk = (const float*)d_in[4];   // (1024,4096)
  const float* Wv = (const float*)d_in[5];   // (1024,4096)
  const float* Wo = (const float*)d_in[6];   // (4096,4096)
  float* out = (float*)d_out;                // (2048,4096) f32

  // workspace carve-up (bf16 buffers), ~168 MB total
  char* ws = (char*)d_ws;
  size_t off = 0;
  auto alloc = [&](size_t elems) {
    __bf16* p = (__bf16*)(ws + off);
    off += ((elems * sizeof(__bf16) + 255) & ~(size_t)255);
    return p;
  };
  __bf16* xb    = alloc((size_t)SEQ * HIDDEN);       // x bf16
  __bf16* wqkvb = alloc((size_t)QKVN * HIDDEN);      // [Wq;Wk;Wv] bf16
  __bf16* wob   = alloc((size_t)HIDDEN * HIDDEN);    // Wo bf16
  __bf16* qkvb  = alloc((size_t)SEQ * QKVN);         // fused qkv output
  __bf16* qrope = alloc((size_t)NH * SEQ * HD);      // q  [H][S][D]
  __bf16* krope = alloc((size_t)NKV * SEQ * HD);     // k  [HKV][S][D]
  __bf16* vtb   = alloc((size_t)NKV * HD * SEQ);     // v^T[HKV][D][S]
  __bf16* attnb = alloc((size_t)SEQ * NH * HD);      // attn out bf16
  (void)ws_size;

  // 1) converts
  cvt_f32_bf16<<<4096, 256, 0, stream>>>(hs, xb, (long long)SEQ * HIDDEN);
  cvt_f32_bf16<<<8192, 256, 0, stream>>>(Wq, wqkvb, (long long)HIDDEN * HIDDEN);
  cvt_f32_bf16<<<2048, 256, 0, stream>>>(Wk, wqkvb + (size_t)HIDDEN * HIDDEN,
                                         (long long)NKV * HD * HIDDEN);
  cvt_f32_bf16<<<2048, 256, 0, stream>>>(Wv, wqkvb + (size_t)(HIDDEN + NKV * HD) * HIDDEN,
                                         (long long)NKV * HD * HIDDEN);
  cvt_f32_bf16<<<8192, 256, 0, stream>>>(Wo, wob, (long long)HIDDEN * HIDDEN);

  // 2) fused QKV projection: qkv[2048,6144] = x @ [Wq;Wk;Wv]^T
  gemm_bf16<<<dim3(SEQ / 32, QKVN / 256), 128, 0, stream>>>(
      xb, wqkvb, qkvb, nullptr, SEQ, QKVN, HIDDEN);

  // 3) RoPE + relayout (+ D^-0.5 folded into q, V transposed)
  rope_layout<<<(SEQ * NH * 64) / 256, 256, 0, stream>>>(qkvb, pos, qrope, krope, vtb);

  // 4) causal GQA flash attention
  attn_kernel<<<dim3(SEQ / 16, NH), 32, 0, stream>>>(qrope, krope, vtb, attnb);

  // 5) output projection: out[2048,4096] = attn @ Wo^T  (f32 out)
  gemm_bf16<<<dim3(SEQ / 32, HIDDEN / 256), 128, 0, stream>>>(
      attnb, wob, nullptr, out, SEQ, HIDDEN, HIDDEN);
}
